// HybridMOFModel_89859305767804
// MI455X (gfx1250) — matrix operand order repaired and checked by reading, not compiled, not run on hardware
//
#include <hip/hip_runtime.h>
#include <hip/hip_bf16.h>
#include <math.h>

typedef __attribute__((ext_vector_type(16))) _Float16 v16h;
typedef __attribute__((ext_vector_type(8)))  float    v8f;

#define NN 50000
#define EE 800000
#define BB 512
#define HH 128
#define RBFN 50
#define CHEMIN 145
#define QIN 8
#define QOUT 64

// silu with hardware reciprocal (v_rcp_f32) instead of IEEE div chain
__device__ __forceinline__ float siluf(float x){
  return x * __builtin_amdgcn_rcpf(1.f + __expf(-x));
}
__device__ __forceinline__ float geluf(float x){ return 0.5f * x * (1.f + erff(x * 0.7071067811865475f)); }

// ---- WMMA fragment loaders (16x16x32 f16, wave32) ----
// A (16xK tile, row-major in LDS, stride ldk): lane -> row M=lane&15;
// halves j<8 -> K = k0 + hi*8 + j ; j>=8 -> K = k0 + hi*8 + 16 + (j-8)
__device__ __forceinline__ v16h load_A(const _Float16* base, int ldk, int k0, int lane){
  const int row = lane & 15, hi = lane >> 4;
  const _Float16* p = base + row*ldk + k0 + hi*8;
  v16h a;
  #pragma unroll
  for (int j = 0; j < 8; ++j) a[j] = p[j];
  #pragma unroll
  for (int j = 0; j < 8; ++j) a[8+j] = p[16+j];
  return a;
}
// B (KxN tile, row-major [k][n], stride ldn), columns n0..n0+15:
// lane -> col N = n0 + (lane&15); halves j<8 -> K = k0 + hi*8 + j ; j>=8 -> K = k0 + hi*8 + 16 + (j-8)
__device__ __forceinline__ v16h load_B(const _Float16* base, int ldn, int k0, int n0, int lane){
  const int col = lane & 15, hi = lane >> 4;
  const _Float16* p = base + (k0 + hi*8)*ldn + n0 + col;
  v16h b;
  #pragma unroll
  for (int j = 0; j < 8; ++j) b[j] = p[j*ldn];
  #pragma unroll
  for (int j = 0; j < 8; ++j) b[8+j] = p[(16+j)*ldn];
  return b;
}
#define WMMA_F16(A,B,C) __builtin_amdgcn_wmma_f32_16x16x32_f16(false,(A),false,(B),(short)0,(C),false,false)

// ================= node embed: h = silu(nf @ ne_w^T + ne_b) =================
__global__ void k_node_embed(const float* __restrict__ nf, const float* __restrict__ w,
                             const float* __restrict__ b, float* __restrict__ h){
  int idx = blockIdx.x*blockDim.x + threadIdx.x;
  if (idx >= NN*HH) return;
  int n = idx >> 7, j = idx & 127;
  float a = b[j];
  #pragma unroll
  for (int k = 0; k < 3; ++k) a += nf[n*3+k]*w[j*3+k];
  h[idx] = siluf(a);
}

// ===== fused SchNet edge interaction: rbf-gen -> MLP(WMMA) -> gather*W -> scatter-add =====
__global__ __launch_bounds__(256) void k_edge_interaction(
    const float* __restrict__ dist, const int* __restrict__ eidx,
    const float* __restrict__ h,
    const float* __restrict__ f1w, const float* __restrict__ f1b,
    const float* __restrict__ f2w, const float* __restrict__ f2b,
    float* __restrict__ agg)
{
  __shared__ _Float16 sW1[64*HH];    // [k][n] = f1w[n][k], zero-padded K 50->64
  __shared__ _Float16 sW2[HH*HH];    // [k][n] = f2w[n][k]
  __shared__ float    sB1[HH], sB2[HH];
  __shared__ _Float16 sRbf[16*64];   // 16 edges x 64 K
  __shared__ _Float16 sX1[16*HH];    // silu(gemm1) in f16
  __shared__ int      sSrc[16], sDst[16];

  const int tid = threadIdx.x;
  const int lane = tid & 31, wave = tid >> 5;
  const int n0 = wave * 16;

  for (int i = tid; i < 64*HH; i += 256){
    int k = i >> 7, n = i & 127;
    sW1[i] = (k < RBFN) ? (_Float16)f1w[n*RBFN + k] : (_Float16)0.f;
  }
  for (int i = tid; i < HH*HH; i += 256){
    int k = i >> 7, n = i & 127;
    sW2[i] = (_Float16)f2w[n*HH + k];
  }
  for (int i = tid; i < HH; i += 256){ sB1[i] = f1b[i]; sB2[i] = f2b[i]; }
  __syncthreads();

  // B fragments are tile-invariant: hoist out of the persistent loop
  const v16h b10 = load_B(sW1, HH, 0,  n0, lane);
  const v16h b11 = load_B(sW1, HH, 32, n0, lane);
  const v16h b20 = load_B(sW2, HH, 0,  n0, lane);
  const v16h b21 = load_B(sW2, HH, 32, n0, lane);
  const v16h b22 = load_B(sW2, HH, 64, n0, lane);
  const v16h b23 = load_B(sW2, HH, 96, n0, lane);

  const int nTiles = EE / 16;
  const int tileStride = gridDim.x;
  for (int t = blockIdx.x; t < nTiles; t += tileStride){
    const int e0 = t * 16;
    // prefetch next tile's edge data (global_prefetch_b8)
    {
      int tn = t + tileStride;
      if (tn < nTiles && tid == 0){
        __builtin_prefetch(&dist[tn*16], 0, 1);
        __builtin_prefetch(&eidx[tn*16], 0, 1);
        __builtin_prefetch(&eidx[EE + tn*16], 0, 1);
      }
    }
    for (int i = tid; i < 16*64; i += 256){
      int m = i >> 6, k = i & 63;
      float d  = dist[e0 + m];
      float ck = (float)k * (6.0f / 49.0f);
      float df = d - ck;
      sRbf[i] = (k < RBFN) ? (_Float16)__expf(-10.f * df * df) : (_Float16)0.f;
    }
    if (tid < 16){ sSrc[tid] = eidx[e0 + tid]; sDst[tid] = eidx[EE + e0 + tid]; }
    __syncthreads();

    // GEMM1: (16x64 rbf) @ (64x128 W1) -> per-wave 16x16 tile
    v8f c1 = {};
    c1 = WMMA_F16(load_A(sRbf, 64, 0,  lane), b10, c1);
    c1 = WMMA_F16(load_A(sRbf, 64, 32, lane), b11, c1);

    {
      const int col = lane & 15, hi = lane >> 4;
      const float bias = sB1[n0 + col];
      #pragma unroll
      for (int v = 0; v < 8; ++v){
        int m = v + 8*hi;                           // C/D: VGPR v -> row m
        sX1[m*HH + n0 + col] = (_Float16)siluf(c1[v] + bias);
      }
    }
    __syncthreads();

    // GEMM2: (16x128 X1) @ (128x128 W2)
    v8f c2 = {};
    c2 = WMMA_F16(load_A(sX1, HH, 0,  lane), b20, c2);
    c2 = WMMA_F16(load_A(sX1, HH, 32, lane), b21, c2);
    c2 = WMMA_F16(load_A(sX1, HH, 64, lane), b22, c2);
    c2 = WMMA_F16(load_A(sX1, HH, 96, lane), b23, c2);

    // message = h[src] * W ; scatter-add into agg[dst]  (h, agg are L2-resident)
    {
      const int col = lane & 15, hi = lane >> 4;
      const int n = n0 + col;
      const float bias = sB2[n];
      #pragma unroll
      for (int v = 0; v < 8; ++v){
        int m = v + 8*hi;
        int s = sSrc[m], dnode = sDst[m];
        float Wv  = c2[v] + bias;
        float msg = h[s*HH + n] * Wv;
        atomicAdd(&agg[dnode*HH + n], msg);
      }
    }
    __syncthreads();
  }
}

// ============ node update: h = h + silu(agg @ lw^T + lb), WMMA ============
__global__ __launch_bounds__(256) void k_node_update(
    float* __restrict__ h, const float* __restrict__ agg,
    const float* __restrict__ lw, const float* __restrict__ lb)
{
  __shared__ _Float16 sW[HH*HH];
  __shared__ float    sB[HH];
  __shared__ _Float16 sA[16*HH];

  const int tid = threadIdx.x;
  const int lane = tid & 31, wave = tid >> 5;
  const int n0 = wave * 16;

  for (int i = tid; i < HH*HH; i += 256){
    int k = i >> 7, n = i & 127;
    sW[i] = (_Float16)lw[n*HH + k];
  }
  for (int i = tid; i < HH; i += 256) sB[i] = lb[i];
  __syncthreads();

  const v16h b0 = load_B(sW, HH, 0,  n0, lane);
  const v16h b1 = load_B(sW, HH, 32, n0, lane);
  const v16h b2 = load_B(sW, HH, 64, n0, lane);
  const v16h b3 = load_B(sW, HH, 96, n0, lane);

  const int nTiles = (NN + 15) / 16;
  for (int t = blockIdx.x; t < nTiles; t += gridDim.x){
    const int r0 = t * 16;
    for (int i = tid; i < 16*HH; i += 256){
      int m = i >> 7, n = i & 127;
      int node = r0 + m;
      sA[i] = (node < NN) ? (_Float16)agg[node*HH + n] : (_Float16)0.f;
    }
    __syncthreads();

    v8f c = {};
    c = WMMA_F16(load_A(sA, HH, 0,  lane), b0, c);
    c = WMMA_F16(load_A(sA, HH, 32, lane), b1, c);
    c = WMMA_F16(load_A(sA, HH, 64, lane), b2, c);
    c = WMMA_F16(load_A(sA, HH, 96, lane), b3, c);

    {
      const int col = lane & 15, hi = lane >> 4;
      const int n = n0 + col;
      const float bias = sB[n];
      #pragma unroll
      for (int v = 0; v < 8; ++v){
        int node = r0 + v + 8*hi;
        if (node < NN) h[node*HH + n] += siluf(c[v] + bias);
      }
    }
    __syncthreads();
  }
}

// ============ attention-pool gate: gate[n] = g2 @ silu(g1 @ h[n]) ============
__global__ __launch_bounds__(256) void k_gate(const float* __restrict__ h,
    const float* __restrict__ g1w, const float* __restrict__ g1b,
    const float* __restrict__ g2w, const float* __restrict__ g2b,
    float* __restrict__ gate)
{
  __shared__ float sW[64*HH];
  __shared__ float sB1[64], sW2[64];
  for (int i = threadIdx.x; i < 64*HH; i += 256) sW[i] = g1w[i];
  for (int i = threadIdx.x; i < 64; i += 256){ sB1[i] = g1b[i]; sW2[i] = g2w[i]; }
  __syncthreads();
  for (int n = blockIdx.x*256 + threadIdx.x; n < NN; n += gridDim.x*256){
    const float* hr = h + n*HH;
    float acc = g2b[0];
    for (int j = 0; j < 64; ++j){
      float s = sB1[j];
      #pragma unroll 4
      for (int k = 0; k < HH; ++k) s += sW[j*HH+k]*hr[k];
      acc += sW2[j]*siluf(s);
    }
    gate[n] = acc;
  }
}

// order-preserving float <-> uint map for atomicMax on floats
__device__ __forceinline__ unsigned fmapu(float f){
  unsigned u = __float_as_uint(f);
  return (u & 0x80000000u) ? ~u : (u | 0x80000000u);
}
__device__ __forceinline__ float fumap(unsigned u){
  return __uint_as_float((u & 0x80000000u) ? (u & 0x7fffffffu) : ~u);
}

__global__ void k_pool_init(unsigned* __restrict__ gmaxU){
  int i = blockIdx.x*blockDim.x + threadIdx.x;
  if (i < BB) gmaxU[i] = 0x007FFFFFu;    // fmapu(-inf)
}
__global__ void k_gate_max(const float* __restrict__ gate, const int* __restrict__ batch,
                           unsigned* __restrict__ gmaxU){
  int n = blockIdx.x*blockDim.x + threadIdx.x;
  if (n < NN) atomicMax(&gmaxU[batch[n]], fmapu(gate[n]));
}
__global__ void k_gate_exps(const float* __restrict__ gate, const int* __restrict__ batch,
                            const unsigned* __restrict__ gmaxU,
                            float* __restrict__ ge, float* __restrict__ gs){
  int n = blockIdx.x*blockDim.x + threadIdx.x;
  if (n >= NN) return;
  int b = batch[n];
  float e = __expf(gate[n] - fumap(gmaxU[b]));
  ge[n] = e;
  atomicAdd(&gs[b], e);
}
__global__ void k_pool_weighted(const float* __restrict__ h, const int* __restrict__ batch,
                                const float* __restrict__ ge, const float* __restrict__ gs,
                                float* __restrict__ hg){
  int idx = blockIdx.x*blockDim.x + threadIdx.x;
  if (idx >= NN*HH) return;
  int n = idx >> 7, j = idx & 127;
  int b = batch[n];
  float w = ge[n] / (gs[b] + 1e-8f);
  atomicAdd(&hg[b*HH + j], w * h[idx]);
}

// ============ generic linear + LayerNorm + GELU (block per sample, Out==blockDim) ============
__global__ void k_lin_ln_gelu(const float* __restrict__ x, const float* __restrict__ w,
                              const float* __restrict__ bias, const float* __restrict__ gamma,
                              const float* __restrict__ beta, float* __restrict__ out,
                              int In, int Out)
{
  __shared__ float red[256];
  const int b = blockIdx.x, j = threadIdx.x, T = blockDim.x;
  const float* xr = x + (long)b*In;
  const float* wr = w + (long)j*In;
  float acc = bias[j];
  for (int k = 0; k < In; ++k) acc += wr[k]*xr[k];
  red[j] = acc; __syncthreads();
  for (int s = T>>1; s > 0; s >>= 1){ if (j < s) red[j] += red[j+s]; __syncthreads(); }
  float mean = red[0] / (float)T; __syncthreads();
  float d = acc - mean;
  red[j] = d*d; __syncthreads();
  for (int s = T>>1; s > 0; s >>= 1){ if (j < s) red[j] += red[j+s]; __syncthreads(); }
  float var = red[0] / (float)T;
  float y = d * rsqrtf(var + 1e-5f) * gamma[j] + beta[j];
  out[(long)b*Out + j] = geluf(y);
}

// generic linear + optional activation (thread per output element)
__global__ void k_lin_act(const float* __restrict__ x, const float* __restrict__ w,
                          const float* __restrict__ bias, float* __restrict__ out,
                          int In, int Out, int total, int act)
{
  int idx = blockIdx.x*blockDim.x + threadIdx.x;
  if (idx >= total) return;
  int b = idx / Out, j = idx - b*Out;
  const float* xr = x + (long)b*In;
  const float* wr = w + (long)j*In;
  float acc = bias[j];
  for (int k = 0; k < In; ++k) acc += wr[k]*xr[k];
  if (act == 1) acc = geluf(acc);
  out[idx] = acc;
}

__global__ void k_qmask(const float* __restrict__ qn, const unsigned char* __restrict__ mask,
                        const float* __restrict__ qmiss, float* __restrict__ q){
  int idx = blockIdx.x*blockDim.x + threadIdx.x;
  if (idx >= BB*QOUT) return;
  int b = idx / QOUT, j = idx - b*QOUT;
  q[idx] = mask[b] ? qn[idx] : qmiss[j];
}

// cross-attention with single KV token: softmax(len-1)==1 -> fused = LN(lin(V)+c), V=lin(g)
__global__ __launch_bounds__(128) void k_fusion(const float* __restrict__ g, const float* __restrict__ c,
    const float* __restrict__ fvw, const float* __restrict__ fvb,
    const float* __restrict__ fow, const float* __restrict__ fob,
    const float* __restrict__ flg, const float* __restrict__ flb,
    float* __restrict__ fused)
{
  __shared__ float vv[HH];
  __shared__ float red[HH];
  const int b = blockIdx.x, j = threadIdx.x;
  const float* gr = g + b*HH;
  float a = fvb[j];
  for (int k = 0; k < HH; ++k) a += fvw[j*HH+k]*gr[k];
  vv[j] = a; __syncthreads();
  float o = fob[j] + c[b*HH + j];
  for (int k = 0; k < HH; ++k) o += fow[j*HH+k]*vv[k];
  red[j] = o; __syncthreads();
  for (int s = 64; s > 0; s >>= 1){ if (j < s) red[j] += red[j+s]; __syncthreads(); }
  float mean = red[0] / (float)HH; __syncthreads();
  float d = o - mean;
  red[j] = d*d; __syncthreads();
  for (int s = 64; s > 0; s >>= 1){ if (j < s) red[j] += red[j+s]; __syncthreads(); }
  float var = red[0] / (float)HH;
  fused[b*HH + j] = d * rsqrtf(var + 1e-5f) * flg[j] + flb[j];
}

__global__ void k_concat(const float* __restrict__ fused, const float* __restrict__ q,
                         float* __restrict__ xcat){
  int idx = blockIdx.x*blockDim.x + threadIdx.x;
  if (idx >= BB*(HH+QOUT)) return;
  int b = idx / (HH+QOUT), j = idx - b*(HH+QOUT);
  xcat[idx] = (j < HH) ? fused[b*HH + j] : q[b*QOUT + (j - HH)];
}

// =============================== host side ===============================
extern "C" void kernel_launch(void* const* d_in, const int* in_sizes, int n_in,
                              void* d_out, int out_size, void* d_ws, size_t ws_size,
                              hipStream_t stream) {
  (void)in_sizes; (void)n_in; (void)out_size; (void)ws_size;
  const float* nf     = (const float*)d_in[0];
  const int*   eidx   = (const int*)  d_in[1];
  const float* dist   = (const float*)d_in[2];
  const int*   batch  = (const int*)  d_in[3];
  const float* chemx  = (const float*)d_in[4];
  const float* qx     = (const float*)d_in[5];
  const unsigned char* qmask = (const unsigned char*)d_in[6];
  const int P = 7;  // params start (dict insertion order)
  const float* ne_w = (const float*)d_in[P+0];
  const float* ne_b = (const float*)d_in[P+1];
  // layers l: f1w,f1b,f2w,f2b,lw,lb at P+2+6l..
  const float* g1w = (const float*)d_in[P+20];
  const float* g1b = (const float*)d_in[P+21];
  const float* g2w = (const float*)d_in[P+22];
  const float* g2b = (const float*)d_in[P+23];
  const float* pw  = (const float*)d_in[P+24];
  const float* pb  = (const float*)d_in[P+25];
  const float* plg = (const float*)d_in[P+26];
  const float* plb = (const float*)d_in[P+27];
  const float* c1w = (const float*)d_in[P+28];
  const float* c1b = (const float*)d_in[P+29];
  const float* cl1g= (const float*)d_in[P+30];
  const float* cl1b= (const float*)d_in[P+31];
  const float* c2w = (const float*)d_in[P+32];
  const float* c2b = (const float*)d_in[P+33];
  const float* cl2g= (const float*)d_in[P+34];
  const float* cl2b= (const float*)d_in[P+35];
  const float* q1w = (const float*)d_in[P+36];
  const float* q1b = (const float*)d_in[P+37];
  const float* q2w = (const float*)d_in[P+38];
  const float* q2b = (const float*)d_in[P+39];
  const float* qmiss=(const float*)d_in[P+40];
  const float* fvw = (const float*)d_in[P+45];
  const float* fvb = (const float*)d_in[P+46];
  const float* fow = (const float*)d_in[P+47];
  const float* fob = (const float*)d_in[P+48];
  const float* flg = (const float*)d_in[P+49];
  const float* flb = (const float*)d_in[P+50];
  const float* h1w = (const float*)d_in[P+51];
  const float* h1b = (const float*)d_in[P+52];
  const float* hlg = (const float*)d_in[P+53];
  const float* hlb = (const float*)d_in[P+54];
  const float* h2w = (const float*)d_in[P+55];
  const float* h2b = (const float*)d_in[P+56];
  const float* h3w = (const float*)d_in[P+57];
  const float* h3b = (const float*)d_in[P+58];

  // workspace layout (floats)
  float* W = (float*)d_ws;
  float* h     = W;                  // N*H
  float* agg   = h    + (size_t)NN*HH;
  float* gate  = agg  + (size_t)NN*HH;
  float* ge    = gate + NN;
  unsigned* gmaxU = (unsigned*)(ge + NN);   // B
  float* gs    = (float*)(gmaxU + BB);
  float* hg    = gs   + BB;          // B*H
  float* gvec  = hg   + (size_t)BB*HH;
  float* cmid  = gvec + (size_t)BB*HH;      // B*256
  float* cvec  = cmid + (size_t)BB*256;     // B*128
  float* qh    = cvec + (size_t)BB*HH;      // B*64
  float* qn    = qh   + (size_t)BB*64;
  float* qvec  = qn   + (size_t)BB*64;
  float* fused = qvec + (size_t)BB*64;      // B*128
  float* xcat  = fused+ (size_t)BB*HH;      // B*192
  float* x1    = xcat + (size_t)BB*(HH+QOUT); // B*256
  float* x2    = x1   + (size_t)BB*256;     // B*128

  // ---- GNN branch ----
  k_node_embed<<<(NN*HH + 255)/256, 256, 0, stream>>>(nf, ne_w, ne_b, h);
  for (int l = 0; l < 3; ++l){
    const float* f1w = (const float*)d_in[P+2+6*l+0];
    const float* f1b = (const float*)d_in[P+2+6*l+1];
    const float* f2w = (const float*)d_in[P+2+6*l+2];
    const float* f2b = (const float*)d_in[P+2+6*l+3];
    const float* lw  = (const float*)d_in[P+2+6*l+4];
    const float* lb  = (const float*)d_in[P+2+6*l+5];
    hipMemsetAsync(agg, 0, (size_t)NN*HH*sizeof(float), stream);
    k_edge_interaction<<<2048, 256, 0, stream>>>(dist, eidx, h, f1w, f1b, f2w, f2b, agg);
    k_node_update<<<1024, 256, 0, stream>>>(h, agg, lw, lb);
  }
  // ---- attention pool ----
  k_gate<<<256, 256, 0, stream>>>(h, g1w, g1b, g2w, g2b, gate);
  k_pool_init<<<(BB + 255)/256, 256, 0, stream>>>(gmaxU);
  hipMemsetAsync(gs, 0, BB*sizeof(float), stream);
  hipMemsetAsync(hg, 0, (size_t)BB*HH*sizeof(float), stream);
  k_gate_max <<<(NN + 255)/256, 256, 0, stream>>>(gate, batch, gmaxU);
  k_gate_exps<<<(NN + 255)/256, 256, 0, stream>>>(gate, batch, gmaxU, ge, gs);
  k_pool_weighted<<<(NN*HH + 255)/256, 256, 0, stream>>>(h, batch, ge, gs, hg);
  k_lin_ln_gelu<<<BB, 128, 0, stream>>>(hg, pw, pb, plg, plb, gvec, HH, HH);
  // ---- chemical branch ----
  k_lin_ln_gelu<<<BB, 256, 0, stream>>>(chemx, c1w, c1b, cl1g, cl1b, cmid, CHEMIN, 256);
  k_lin_ln_gelu<<<BB, 128, 0, stream>>>(cmid, c2w, c2b, cl2g, cl2b, cvec, 256, 128);
  // ---- quantum branch ----
  k_lin_act<<<(BB*64 + 255)/256, 256, 0, stream>>>(qx, q1w, q1b, qh, QIN, 64, BB*64, 1);
  k_lin_act<<<(BB*QOUT + 255)/256, 256, 0, stream>>>(qh, q2w, q2b, qn, 64, QOUT, BB*QOUT, 1);
  k_qmask<<<(BB*QOUT + 255)/256, 256, 0, stream>>>(qn, qmask, qmiss, qvec);
  // ---- fusion (seq-len-1 attention collapses to V) ----
  k_fusion<<<BB, 128, 0, stream>>>(gvec, cvec, fvw, fvb, fow, fob, flg, flb, fused);
  // ---- head ----
  k_concat<<<(BB*(HH+QOUT) + 255)/256, 256, 0, stream>>>(fused, qvec, xcat);
  k_lin_ln_gelu<<<BB, 256, 0, stream>>>(xcat, h1w, h1b, hlg, hlb, x1, HH+QOUT, 256);
  k_lin_act<<<(BB*128 + 255)/256, 256, 0, stream>>>(x1, h2w, h2b, x2, 256, 128, BB*128, 1);
  k_lin_act<<<(BB + 255)/256, 256, 0, stream>>>(x2, h3w, h3b, (float*)d_out, 128, 1, BB, 0);
}
